// GraphProcessor_39247411151377
// MI455X (gfx1250) — compile-verified, matrix operand order checked by reading, not run on hardware
//
#include <hip/hip_runtime.h>

// ---------------------------------------------------------------------------
// GNN message-passing block on MI455X (gfx1250), bf16 WMMA + f32 accumulate.
// Only the LAST of the 4 parameter blocks affects the output (the reference's
// scan never feeds the carry back into the step body), so we compute exactly
// one edge-MLP + scatter-sum + node-MLP with block-3 weights.
//
// Roofline: ~138 GFLOP, ~1 GB HBM traffic -> ~43us memory floor @ 23.3 TB/s;
// bf16 WMMA keeps compute in the same ballpark. Each wave register-blocks a
// 32x64 output tile (2 row tiles x 4 col tiles) so each B (weight) fragment
// feeds two WMMAs, halving L2/WGP$ weight traffic vs 16x128 tiling.
// ---------------------------------------------------------------------------

#define NNODES 50000
#define NEDGES 800000
#define DF     128
#define BLK    3
#define EPSLN  1e-5f

typedef __attribute__((ext_vector_type(16))) __bf16 bf16x16;
typedef __attribute__((ext_vector_type(8)))  float  floatx8;

__device__ __forceinline__ unsigned short f2bf(float f) {
  unsigned int u = __float_as_uint(f);
  u += 0x7FFFu + ((u >> 16) & 1u);          // round-to-nearest-even
  return (unsigned short)(u >> 16);
}

// Load a 16x32 bf16 WMMA operand fragment from a row-major [rows][stride]
// bf16 array.  Contiguous K order, same for A and B: lanes 0-15 hold row (M or N) = lane, K=k0..k0+15;
// lanes 16-31 hold the same rows, K=k0+16..k0+31.  One 32-byte load per lane.
__device__ __forceinline__ bf16x16 frag_ld(const unsigned short* base, int row0,
                                           int stride, int k0) {
  const int lane = threadIdx.x & 31;
  const int r = row0 + (lane & 15);
  const int k = k0 + ((lane >> 4) << 4);
  return *(const bf16x16*)(base + r * stride + k);
}

// One GEMM layer: [128 rows x (KT*32)] @ [(KT*32) x 128] + bias (+ReLU).
// Wave w owns rows (w>>1)*32..+31 and col tiles (w&1)*4..+3; each B fragment
// is reused by two WMMAs (rows m0 and m0+16).
template <int KT, bool RELU, bool OUT_BF16>
__device__ __forceinline__ void mlp_layer(const unsigned short* sA, int strideA,
                                          const unsigned short* wt, int strideW,
                                          const float* __restrict__ bias,
                                          unsigned short* outU, int strideU,
                                          float* outF, int strideF) {
  const int tid   = threadIdx.x;
  const int lane  = tid & 31;
  const int wv    = tid >> 5;
  const int m0    = (wv >> 1) * 32;           // 32-row stripe
  const int c0    = (wv & 1) * 4;             // 4 col tiles (64 cols)
  const int nlane = lane & 15;
  const int mhalf = (lane >> 4) * 8;
  for (int ct = c0; ct < c0 + 4; ++ct) {
    floatx8 acc0 = {}, acc1 = {};
#pragma unroll
    for (int ks = 0; ks < KT; ++ks) {
      bf16x16 b  = frag_ld(wt, ct * 16, strideW, ks * 32);
      bf16x16 a0 = frag_ld(sA, m0,      strideA, ks * 32);
      bf16x16 a1 = frag_ld(sA, m0 + 16, strideA, ks * 32);
      acc0 = __builtin_amdgcn_wmma_f32_16x16x32_bf16(false, a0, false, b,
                                                     (short)0, acc0, false, false);
      acc1 = __builtin_amdgcn_wmma_f32_16x16x32_bf16(false, a1, false, b,
                                                     (short)0, acc1, false, false);
    }
    const float bv = bias[ct * 16 + nlane];
    const int   nn = ct * 16 + nlane;
#pragma unroll
    for (int r = 0; r < 8; ++r) {
      float v0 = acc0[r] + bv;
      float v1 = acc1[r] + bv;
      if (RELU) { v0 = fmaxf(v0, 0.0f); v1 = fmaxf(v1, 0.0f); }
      const int mm0 = m0 + mhalf + r;         // C layout: VGPR r -> M = r (+8)
      const int mm1 = mm0 + 16;
      if (OUT_BF16) {
        outU[mm0 * strideU + nn] = f2bf(v0);
        outU[mm1 * strideU + nn] = f2bf(v1);
      } else {
        outF[mm0 * strideF + nn] = v0;
        outF[mm1 * strideF + nn] = v1;
      }
    }
  }
}

// --- prep: Wt[n][k] = bf16(W[k][n]) for one weight matrix (block 3 slice) ---
__global__ void wt_prep(const float* __restrict__ W, unsigned short* __restrict__ Wt,
                        int K) {
  int i = blockIdx.x * blockDim.x + threadIdx.x;
  if (i < 128 * K) {
    int n = i / K, k = i % K;
    Wt[i] = f2bf(W[(long)k * 128 + n]);
  }
}

__global__ void zero_f32(float* __restrict__ p, int n) {
  int i = blockIdx.x * blockDim.x + threadIdx.x;
  if (i < n) p[i] = 0.0f;
}

// ---------------------------------------------------------------------------
// Edge kernel: 128 edges per workgroup, 256 threads (8 wave32).
// LDS: sIn 128x400 bf16 (102400B) | sH1 128x144 bf16 | sH2 128x144 bf16 |
//      src/dst index arrays.  sF (f32 128x132) aliases sIn for layer 3.
// ---------------------------------------------------------------------------
__global__ __launch_bounds__(256) void edge_kernel(
    const float* __restrict__ x, const float* __restrict__ edge_attr,
    const int* __restrict__ edge_index,
    const unsigned short* __restrict__ wt1, const unsigned short* __restrict__ wt2,
    const unsigned short* __restrict__ wt3,
    const float* __restrict__ b1, const float* __restrict__ b2,
    const float* __restrict__ b3, const float* __restrict__ g,
    const float* __restrict__ beta,
    float* __restrict__ out_e, float* __restrict__ agg) {
  extern __shared__ char smem[];
  unsigned short* sIn = (unsigned short*)smem;                  // [128][400]
  unsigned short* sH1 = (unsigned short*)(smem + 102400);       // [128][144]
  unsigned short* sH2 = (unsigned short*)(smem + 139264);       // [128][144]
  int*   sSrc = (int*)(smem + 176128);                          // [128]
  int*   sDst = (int*)(smem + 176640);                          // [128]
  float* sF   = (float*)smem;                                   // [128][132]

  const int  tid   = threadIdx.x;
  const long eBase = (long)blockIdx.x * 128;

  if (tid < 128) sSrc[tid] = edge_index[eBase + tid];
  else           sDst[tid - 128] = edge_index[(long)NEDGES + eBase + (tid - 128)];
  __syncthreads();

  // Gather + f32->bf16: e_in = [x[src] | x[dst] | edge_attr], 96 float4/row.
  for (int idx = tid; idx < 128 * 96; idx += 256) {
    const int e = idx / 96, c = idx % 96;
    const float* sp;
    if (c < 32)      sp = x + (long)sSrc[e] * DF + c * 4;
    else if (c < 64) sp = x + (long)sDst[e] * DF + (c - 32) * 4;
    else             sp = edge_attr + (eBase + e) * DF + (c - 64) * 4;
    const float4 v = *(const float4*)sp;
    unsigned short* dp = sIn + e * 400 + c * 4;
    dp[0] = f2bf(v.x); dp[1] = f2bf(v.y); dp[2] = f2bf(v.z); dp[3] = f2bf(v.w);
  }
  __syncthreads();

  mlp_layer<12, true,  true >(sIn, 400, wt1, 384, b1, sH1, 144, nullptr, 0);
  __syncthreads();
  mlp_layer<4,  true,  true >(sH1, 144, wt2, 128, b2, sH2, 144, nullptr, 0);
  __syncthreads();
  mlp_layer<4,  false, false>(sH2, 144, wt3, 128, b3, nullptr, 0, sF, 132);
  __syncthreads();

  // LayerNorm + residual write + atomic scatter-sum (one edge per thread).
  if (tid < 128) {
    float mu = 0.0f;
    for (int d = 0; d < 128; ++d) mu += sF[tid * 132 + d];
    mu *= (1.0f / 128.0f);
    float var = 0.0f;
    for (int d = 0; d < 128; ++d) {
      const float t = sF[tid * 132 + d] - mu;
      var += t * t;
    }
    const float rinv = rsqrtf(var * (1.0f / 128.0f) + EPSLN);
    const long e   = eBase + tid;
    const int  dst = sDst[tid];
    for (int d = 0; d < 128; ++d) {
      const float v = (sF[tid * 132 + d] - mu) * rinv * g[d] + beta[d];
      out_e[e * DF + d] = v + edge_attr[e * DF + d];
      unsafeAtomicAdd(&agg[(long)dst * DF + d], v);   // global_atomic_add_f32
    }
  }
}

// ---------------------------------------------------------------------------
// Node kernel: 128 nodes per workgroup.  n_in = [x | agg], K = 256.
// LDS: sIn 128x272 bf16 (69632B) | sH1 | sH2.  sF aliases sIn.
// ---------------------------------------------------------------------------
__global__ __launch_bounds__(256) void node_kernel(
    const float* __restrict__ x, const float* __restrict__ agg,
    const unsigned short* __restrict__ wt1, const unsigned short* __restrict__ wt2,
    const unsigned short* __restrict__ wt3,
    const float* __restrict__ b1, const float* __restrict__ b2,
    const float* __restrict__ b3, const float* __restrict__ g,
    const float* __restrict__ beta, float* __restrict__ out_x) {
  extern __shared__ char smem[];
  unsigned short* sIn = (unsigned short*)smem;                  // [128][272]
  unsigned short* sH1 = (unsigned short*)(smem + 69632);        // [128][144]
  unsigned short* sH2 = (unsigned short*)(smem + 106496);       // [128][144]
  float* sF = (float*)smem;                                     // [128][132]

  const int  tid   = threadIdx.x;
  const long nBase = (long)blockIdx.x * 128;

  for (int idx = tid; idx < 128 * 64; idx += 256) {
    const int r = idx / 64, c = idx % 64;
    long id = nBase + r;
    if (id >= NNODES) id = NNODES - 1;                 // clamp tail (unwritten)
    const float* sp = (c < 32) ? (x + id * DF + c * 4)
                               : (agg + id * DF + (c - 32) * 4);
    const float4 v = *(const float4*)sp;
    unsigned short* dp = sIn + r * 272 + c * 4;
    dp[0] = f2bf(v.x); dp[1] = f2bf(v.y); dp[2] = f2bf(v.z); dp[3] = f2bf(v.w);
  }
  __syncthreads();

  mlp_layer<8, true,  true >(sIn, 272, wt1, 256, b1, sH1, 144, nullptr, 0);
  __syncthreads();
  mlp_layer<4, true,  true >(sH1, 144, wt2, 128, b2, sH2, 144, nullptr, 0);
  __syncthreads();
  mlp_layer<4, false, false>(sH2, 144, wt3, 128, b3, nullptr, 0, sF, 132);
  __syncthreads();

  if (tid < 128) {
    const long node = nBase + tid;
    if (node < NNODES) {
      float mu = 0.0f;
      for (int d = 0; d < 128; ++d) mu += sF[tid * 132 + d];
      mu *= (1.0f / 128.0f);
      float var = 0.0f;
      for (int d = 0; d < 128; ++d) {
        const float t = sF[tid * 132 + d] - mu;
        var += t * t;
      }
      const float rinv = rsqrtf(var * (1.0f / 128.0f) + EPSLN);
      for (int d = 0; d < 128; ++d) {
        const float v = (sF[tid * 132 + d] - mu) * rinv * g[d] + beta[d];
        out_x[node * DF + d] = v + x[node * DF + d];
      }
    }
  }
}

// ---------------------------------------------------------------------------
extern "C" void kernel_launch(void* const* d_in, const int* in_sizes, int n_in,
                              void* d_out, int out_size, void* d_ws, size_t ws_size,
                              hipStream_t stream) {
  (void)in_sizes; (void)n_in; (void)out_size; (void)ws_size;

  const float* x   = (const float*)d_in[0];
  const float* ea  = (const float*)d_in[1];
  const float* ew1 = (const float*)d_in[2]  + (size_t)BLK * 384 * 128;
  const float* eb1 = (const float*)d_in[3]  + (size_t)BLK * 128;
  const float* ew2 = (const float*)d_in[4]  + (size_t)BLK * 128 * 128;
  const float* eb2 = (const float*)d_in[5]  + (size_t)BLK * 128;
  const float* ew3 = (const float*)d_in[6]  + (size_t)BLK * 128 * 128;
  const float* eb3 = (const float*)d_in[7]  + (size_t)BLK * 128;
  const float* eg  = (const float*)d_in[8]  + (size_t)BLK * 128;
  const float* ebt = (const float*)d_in[9]  + (size_t)BLK * 128;
  const float* nw1 = (const float*)d_in[10] + (size_t)BLK * 256 * 128;
  const float* nb1 = (const float*)d_in[11] + (size_t)BLK * 128;
  const float* nw2 = (const float*)d_in[12] + (size_t)BLK * 128 * 128;
  const float* nb2 = (const float*)d_in[13] + (size_t)BLK * 128;
  const float* nw3 = (const float*)d_in[14] + (size_t)BLK * 128 * 128;
  const float* nb3 = (const float*)d_in[15] + (size_t)BLK * 128;
  const float* ng  = (const float*)d_in[16] + (size_t)BLK * 128;
  const float* nbt = (const float*)d_in[17] + (size_t)BLK * 128;
  const int*   ei  = (const int*)d_in[18];

  float* out_x = (float*)d_out;
  float* out_e = (float*)d_out + (size_t)NNODES * DF;

  // Workspace: agg[N,128] f32 (25.6MB, L2-resident) + transposed bf16 weights.
  char* ws = (char*)d_ws;
  float* agg = (float*)ws;
  unsigned short* wtE1 = (unsigned short*)(ws + (size_t)NNODES * DF * 4);
  unsigned short* wtE2 = wtE1 + 128 * 384;
  unsigned short* wtE3 = wtE2 + 128 * 128;
  unsigned short* wtN1 = wtE3 + 128 * 128;
  unsigned short* wtN2 = wtN1 + 128 * 256;
  unsigned short* wtN3 = wtN2 + 128 * 128;

  wt_prep<<<(128 * 384 + 255) / 256, 256, 0, stream>>>(ew1, wtE1, 384);
  wt_prep<<<(128 * 128 + 255) / 256, 256, 0, stream>>>(ew2, wtE2, 128);
  wt_prep<<<(128 * 128 + 255) / 256, 256, 0, stream>>>(ew3, wtE3, 128);
  wt_prep<<<(128 * 256 + 255) / 256, 256, 0, stream>>>(nw1, wtN1, 256);
  wt_prep<<<(128 * 128 + 255) / 256, 256, 0, stream>>>(nw2, wtN2, 128);
  wt_prep<<<(128 * 128 + 255) / 256, 256, 0, stream>>>(nw3, wtN3, 128);

  const int aggN = NNODES * DF;
  zero_f32<<<(aggN + 255) / 256, 256, 0, stream>>>(agg, aggN);

  edge_kernel<<<NEDGES / 128, 256, 177152, stream>>>(
      x, ea, ei, wtE1, wtE2, wtE3, eb1, eb2, eb3, eg, ebt, out_e, agg);

  node_kernel<<<(NNODES + 127) / 128, 256, 143360, stream>>>(
      x, agg, wtN1, wtN2, wtN3, nb1, nb2, nb3, ng, nbt, out_x);
}
